// VectorQuantizer_21019569946729
// MI455X (gfx1250) — matrix operand order repaired and checked by reading, not compiled, not run on hardware
//
#include <hip/hip_runtime.h>
#include <hip/hip_bf16.h>

typedef __attribute__((ext_vector_type(16))) _Float16 v16h;
typedef __attribute__((ext_vector_type(8)))  _Float16 v8h;
typedef __attribute__((ext_vector_type(8)))  float    v8f;

#define KCODES 512
#define DDIM   64
#define HW     4096
#define NPIX   262144          /* 64 * 64 * 64 pixels */
#define ZT_STRIDE 72           /* 144B row stride: 16B aligned for b128 LDS loads */

// ---------------------------------------------------------------------------
// prep: f32 codebook -> f16 copy (row major) + per-code squared norms,
// and zero the global loss accumulator.
// ---------------------------------------------------------------------------
__global__ void vq_prep(const float* __restrict__ cb,
                        _Float16* __restrict__ cb16,
                        float* __restrict__ cbnorm,
                        float* __restrict__ acc) {
  int k = blockIdx.x * blockDim.x + threadIdx.x;
  if (k == 0) acc[0] = 0.0f;
  if (k < KCODES) {
    float s = 0.0f;
    #pragma unroll
    for (int d = 0; d < DDIM; ++d) {
      float v = cb[k * DDIM + d];
      s += v * v;
      cb16[k * DDIM + d] = (_Float16)v;
    }
    cbnorm[k] = s;
  }
}

// ---------------------------------------------------------------------------
// main: per 256-pixel tile: WMMA distance matmul vs all 512 codes,
// per-row argmin, gather z_q, partial loss reduction.
// ---------------------------------------------------------------------------
__global__ __launch_bounds__(256)
void vq_main(const float* __restrict__ ze,
             const float* __restrict__ cb,
             const _Float16* __restrict__ cb16,
             const float* __restrict__ cbnorm,
             float* __restrict__ out,
             float* __restrict__ acc) {
  __shared__ _Float16 zt[256 * ZT_STRIDE];   // 36 KB: z tile, f16, [pixel][d]
  __shared__ int      idx_lds[256];
  __shared__ float    wsum[8];

  const int t  = threadIdx.x;
  const int wg = blockIdx.x;
  // 256 consecutive pixels, all inside one image b (4096 pixels/image, 16 blocks/image)
  const int  b    = wg >> 4;
  const int  hw0  = (wg & 15) << 8;
  const long base = (long)b * (DDIM * HW) + hw0;   // elem addr = base + d*4096 + pixel

  // warm WGP$ with the f16 codebook (global_prefetch_b8); 128B per thread x2
  __builtin_prefetch(cb16 + t * 64, 0, 3);
  __builtin_prefetch(cb16 + 16384 + t * 64, 0, 3);

  // ---- stage 1: b128 loads of z (4 pixels x 1 channel per thread), f16 into LDS
  const int dd = t >> 6;          // channel sub-index 0..3
  const int c4 = (t & 63) * 4;    // 4 contiguous pixels
  #pragma unroll 4
  for (int it = 0; it < 16; ++it) {
    const int d = it * 4 + dd;
    float4 z = *(const float4*)(ze + base + (long)d * HW + c4);
    zt[(c4 + 0) * ZT_STRIDE + d] = (_Float16)z.x;
    zt[(c4 + 1) * ZT_STRIDE + d] = (_Float16)z.y;
    zt[(c4 + 2) * ZT_STRIDE + d] = (_Float16)z.z;
    zt[(c4 + 3) * ZT_STRIDE + d] = (_Float16)z.w;
  }
  __syncthreads();

  // ---- stage 2: wave-level GEMM + argmin
  const int wv   = t >> 5;        // wave id 0..7
  const int lane = t & 31;
  const int half = lane >> 4;     // 0: lanes 0-15, 1: lanes 16-31
  const int nl   = lane & 15;
  const int o    = half * 8;      // A-fragment K sub-offset per ISA layout

  // A fragments: 2 M-tiles (rows wv*32 + mt*16 + nl) x 2 K-chunks (d=0..31, 32..63)
  v16h afrag[2][2];
  #pragma unroll
  for (int mt = 0; mt < 2; ++mt) {
    const int m = wv * 32 + mt * 16 + nl;
    const _Float16* zr = &zt[m * ZT_STRIDE];
    #pragma unroll
    for (int kc = 0; kc < 2; ++kc) {
      const int dbase = kc * 32;
      v8h lo = *(const v8h*)(zr + dbase + o);        // K = dbase+o .. +7
      v8h hi = *(const v8h*)(zr + dbase + 16 + o);   // K = dbase+16+o .. +7
      v16h av;
      #pragma unroll
      for (int i = 0; i < 8; ++i) { av[i] = lo[i]; av[8 + i] = hi[i]; }
      afrag[mt][kc] = av;
    }
  }

  float minval[2][8];
  int   minidx[2][8];
  #pragma unroll
  for (int mt = 0; mt < 2; ++mt)
    #pragma unroll
    for (int j = 0; j < 8; ++j) { minval[mt][j] = 3.4e38f; minidx[mt][j] = 0; }

  for (int tile = 0; tile < KCODES / 16; ++tile) {
    const int code = tile * 16 + nl;                 // C column this lane owns
    // B fragments: two 8-half runs per lane (K = 8h..+7, 16+8h..+7; 16B aligned), L1/L2 resident
    const _Float16* cr = cb16 + code * DDIM + half * 8;
    v16h b0 = __builtin_shufflevector(*(const v8h*)(cr), *(const v8h*)(cr + 16), 0,1,2,3,4,5,6,7,8,9,10,11,12,13,14,15);        // K = 0..31
    v16h b1 = __builtin_shufflevector(*(const v8h*)(cr + 32), *(const v8h*)(cr + 48), 0,1,2,3,4,5,6,7,8,9,10,11,12,13,14,15);   // K = 32..63
    const float cbn = cbnorm[code];

    #pragma unroll
    for (int mt = 0; mt < 2; ++mt) {
      v8f c = {};
      c = __builtin_amdgcn_wmma_f32_16x16x32_f16(false, afrag[mt][0], false, b0,
                                                 (short)0, c, false, false);
      c = __builtin_amdgcn_wmma_f32_16x16x32_f16(false, afrag[mt][1], false, b1,
                                                 (short)0, c, false, false);
      #pragma unroll
      for (int j = 0; j < 8; ++j) {
        // dist = ||e||^2 - 2 z.e  (||z||^2 constant per row: dropped)
        float dist = cbn - 2.0f * c[j];
        if (dist < minval[mt][j]) { minval[mt][j] = dist; minidx[mt][j] = code; }
      }
    }
  }

  // cross-lane min-reduce over the 16 lanes holding each row (tie -> lowest idx)
  #pragma unroll
  for (int mt = 0; mt < 2; ++mt) {
    #pragma unroll
    for (int j = 0; j < 8; ++j) {
      float v = minval[mt][j];
      int   i = minidx[mt][j];
      #pragma unroll
      for (int off = 1; off < 16; off <<= 1) {
        float ov = __shfl_xor(v, off, 32);
        int   oi = __shfl_xor(i, off, 32);
        if (ov < v || (ov == v && oi < i)) { v = ov; i = oi; }
      }
      minidx[mt][j] = i;
    }
  }
  if (nl == 0) {  // lanes 0 and 16 publish the 8 rows each half-wave owns
    #pragma unroll
    for (int mt = 0; mt < 2; ++mt) {
      const int rbase = wv * 32 + mt * 16 + half * 8;
      #pragma unroll
      for (int j = 0; j < 8; ++j) idx_lds[rbase + j] = minidx[mt][j];
    }
  }
  __syncthreads();

  // ---- stage 3: gather f32 code rows, b128 z_q stores, loss vs f16 z in LDS
  int idxs[4];
  #pragma unroll
  for (int i = 0; i < 4; ++i) idxs[i] = idx_lds[c4 + i];

  float lsum = 0.0f;
  #pragma unroll 4
  for (int it = 0; it < 16; ++it) {
    const int d = it * 4 + dd;
    float4 e;
    e.x = cb[idxs[0] * DDIM + d];
    e.y = cb[idxs[1] * DDIM + d];
    e.z = cb[idxs[2] * DDIM + d];
    e.w = cb[idxs[3] * DDIM + d];
    float z0 = (float)zt[(c4 + 0) * ZT_STRIDE + d];
    float z1 = (float)zt[(c4 + 1) * ZT_STRIDE + d];
    float z2 = (float)zt[(c4 + 2) * ZT_STRIDE + d];
    float z3 = (float)zt[(c4 + 3) * ZT_STRIDE + d];
    *(float4*)(out + base + (long)d * HW + c4) = e;
    float d0 = e.x - z0, d1 = e.y - z1, d2 = e.z - z2, d3 = e.w - z3;
    lsum += d0 * d0 + d1 * d1 + d2 * d2 + d3 * d3;
  }
  #pragma unroll
  for (int off = 16; off >= 1; off >>= 1) lsum += __shfl_xor(lsum, off, 32);
  if (lane == 0) wsum[wv] = lsum;
  __syncthreads();
  if (t == 0) {
    float s = 0.0f;
    #pragma unroll
    for (int w = 0; w < 8; ++w) s += wsum[w];
    atomicAdd(acc, s);
  }
}

// ---------------------------------------------------------------------------
// finalize: loss scalars
// ---------------------------------------------------------------------------
__global__ void vq_finalize(const float* __restrict__ acc, float* __restrict__ out) {
  float m = acc[0] * (1.0f / ((float)NPIX * (float)DDIM));
  out[(long)NPIX * DDIM]     = m;           // codebook loss
  out[(long)NPIX * DDIM + 1] = 0.25f * m;   // commitment loss = BETA * same
}

extern "C" void kernel_launch(void* const* d_in, const int* in_sizes, int n_in,
                              void* d_out, int out_size, void* d_ws, size_t ws_size,
                              hipStream_t stream) {
  (void)in_sizes; (void)n_in; (void)out_size; (void)ws_size;
  const float* ze = (const float*)d_in[0];   // (64, 64, 64, 64) f32
  const float* cb = (const float*)d_in[1];   // (512, 64) f32
  float* out = (float*)d_out;

  // workspace layout (32B-aligned f16 codebook for v16h loads)
  char* ws = (char*)d_ws;
  float*    acc    = (float*)ws;                 // @0
  float*    cbnorm = (float*)(ws + 256);         // @256, 2 KB
  _Float16* cb16   = (_Float16*)(ws + 4096);     // @4096, 64 KB

  vq_prep<<<2, 256, 0, stream>>>(cb, cb16, cbnorm, acc);
  vq_main<<<NPIX / 256, 256, 0, stream>>>(ze, cb, cb16, cbnorm, out, acc);
  vq_finalize<<<1, 1, 0, stream>>>(acc, out);
}
